// HDCEncoder_87016037416990
// MI455X (gfx1250) — matrix operand order repaired and checked by reading, not compiled, not run on hardware
//
#include <hip/hip_runtime.h>
#include <hip/hip_bf16.h>

typedef _Float16 f16;
typedef __attribute__((ext_vector_type(4)))  _Float16 v4h;
typedef __attribute__((ext_vector_type(8)))  _Float16 v8h;
typedef __attribute__((ext_vector_type(16))) _Float16 v16h;
typedef __attribute__((ext_vector_type(8)))  float    v8f;

#define NB   4
#define NT   2048
#define ND   1024
#define NROW (NB * NT)      // 8192 output rows

// ---------------------------------------------------------------------------
// helpers
// ---------------------------------------------------------------------------
static __device__ __forceinline__ v16h cat8(v8h lo, v8h hi) {
    return __builtin_shufflevector(lo, hi, 0,1,2,3,4,5,6,7,8,9,10,11,12,13,14,15);
}

static __device__ __forceinline__ v8f wmma16(v16h a, v16h b, v8f c) {
    return __builtin_amdgcn_wmma_f32_16x16x32_f16(
        /*neg_a=*/false, a, /*neg_b=*/false, b,
        /*c_mod=*/(short)0, c, /*reuse_a=*/false, /*reuse_b=*/false);
}

static __device__ __forceinline__ unsigned lds_addr32(const void* p) {
    // LDS aperture bits live in addr[63:32]; LDS offset is the low 32 bits
    return (unsigned)(unsigned long long)p;
}

static __device__ __forceinline__ void async_b128(unsigned ldsb, unsigned gofs,
                                                  unsigned long long base) {
    // GVS mode: mem = SGPR64 + VGPR32, LDS dest byte addr in vdst VGPR
    asm volatile("global_load_async_to_lds_b128 %0, %1, %2"
                 :: "v"(ldsb), "v"(gofs), "s"(base) : "memory");
}

// ---------------------------------------------------------------------------
// Kernel 1: bundle[b,t,d]  (values are multiples of 0.25 in [-1,1] -> exact f16)
//   bundle = v0[d] + v1[d]*v0[d-1] + v2[d]*v1[d-1]*v0[d-2]
//          + v3[d]*v2[d-1]*v1[d-2]*v0[d-3], feature index mod 1024,
//   vj = token_vectors[tokens[b, t-j]] (zero row if t-j < 0), then * 0.25
// ---------------------------------------------------------------------------
__global__ __launch_bounds__(256) void hdc_bundle_kernel(
    const int*   __restrict__ tokens,   // [NB, NT]
    const float* __restrict__ tv,       // [VOCAB, ND]
    f16*         __restrict__ bundleH)  // [NROW, ND]
{
    __shared__ __align__(16) float rows[4][ND];
    const int tid = threadIdx.x;
    const int bt  = blockIdx.x;
    const int b   = bt >> 11;
    const int t   = bt & (NT - 1);

    #pragma unroll
    for (int j = 0; j < 4; ++j) {
        const int tj = t - j;
        float4 val;
        if (tj >= 0) {
            const int tok = tokens[b * NT + tj];
            val = ((const float4*)(tv + (size_t)tok * ND))[tid];
        } else {
            val = make_float4(0.f, 0.f, 0.f, 0.f);
        }
        ((float4*)rows[j])[tid] = val;
    }
    __syncthreads();

    const int d0 = tid * 4;
    v4h outv;
    #pragma unroll
    for (int i = 0; i < 4; ++i) {
        const int d  = d0 + i;
        const int d1 = (d - 1) & (ND - 1);
        const int d2 = (d - 2) & (ND - 1);
        const int d3 = (d - 3) & (ND - 1);
        const float v0a = rows[0][d],  v0b = rows[0][d1];
        const float v0c = rows[0][d2], v0d = rows[0][d3];
        const float v1a = rows[1][d],  v1b = rows[1][d1], v1c = rows[1][d2];
        const float v2a = rows[2][d],  v2b = rows[2][d1];
        const float v3a = rows[3][d];
        const float bun = v0a
                        + v1a * v0b
                        + v2a * v1b * v0c
                        + v3a * v2b * v1c * v0d;
        outv[i] = (f16)(bun * 0.25f);
    }
    *(v4h*)(bundleH + (size_t)bt * ND + d0) = outv;
}

// ---------------------------------------------------------------------------
// Kernel 2: split W into f16 hi/lo, transposed to k-major:
//   B[k][n] = W[n][k];  Bhi + Blo == B to ~f32 precision
// ---------------------------------------------------------------------------
__global__ __launch_bounds__(256) void hdc_wsplit_kernel(
    const float* __restrict__ W,        // [ND(e), ND(d)]
    f16*         __restrict__ WThi,     // [ND(d), ND(e)]
    f16*         __restrict__ WTlo)
{
    const int idx = blockIdx.x * 256 + threadIdx.x;
    const int e = idx >> 10;
    const int d = idx & (ND - 1);
    const float w  = W[idx];
    const f16 hi   = (f16)w;
    const f16 lo   = (f16)(w - (float)hi);
    WThi[d * ND + e] = hi;
    WTlo[d * ND + e] = lo;
}

// ---------------------------------------------------------------------------
// Kernel 3: out[r,n] = sum_k bundle[r,k]*(Bhi+Blo)[k,n] + bias[n]
//   block tile 128(M) x 128(N), 8 waves (4 M x 2 N), wave tile 32x64,
//   K-step 32, async-LDS double buffering (6 async b128 / thread / step).
// ---------------------------------------------------------------------------
#define AS_STRIDE 40                    // 32 + 8 pad (f16 elems)
#define BS_STRIDE 136                   // 128 + 8 pad (f16 elems)
#define AS_TILE   (128 * AS_STRIDE)
#define BS_TILE   (32 * BS_STRIDE)

__global__ __launch_bounds__(256) void hdc_gemm_kernel(
    const f16*   __restrict__ A,     // [NROW, ND]
    const f16*   __restrict__ Bhi,   // [ND(k), ND(n)]
    const f16*   __restrict__ Blo,   // [ND(k), ND(n)]
    const float* __restrict__ bias,  // [ND]
    float*       __restrict__ out)   // [NROW, ND]
{
    __shared__ __align__(16) f16 As  [2 * AS_TILE];
    __shared__ __align__(16) f16 BsH [2 * BS_TILE];
    __shared__ __align__(16) f16 BsL [2 * BS_TILE];

    const int tid  = threadIdx.x;
    const int lane = tid & 31;
    const int wave = tid >> 5;
    const int wm   = wave & 3;          // 0..3 -> 32-row M sub-tile
    const int wn   = wave >> 2;         // 0..1 -> 64-col N sub-tile

    const int mt = blockIdx.x & 63;     // 64 M tiles
    const int nt = blockIdx.x >> 6;     // 8  N tiles
    const int m0 = mt * 128;
    const int n0 = nt * 128;

    const unsigned long long Ab  = (unsigned long long)A;
    const unsigned long long Bhb = (unsigned long long)Bhi;
    const unsigned long long Blb = (unsigned long long)Blo;

    v8f acc[8] = {};                    // acc[mi*4+f]

    // stage one 24KB K-slab (A 128x32, Bhi/Blo 32x128) into buffer `buf`
    auto stage = [&](int k0, int buf) {
        #pragma unroll
        for (int c = 0; c < 2; ++c) {
            const int chunk = tid * 2 + c;          // 0..511
            const int row   = chunk >> 2;           // 0..127
            const int col8  = (chunk & 3) * 8;      // 0,8,16,24
            const unsigned ldsb = lds_addr32(As + buf * AS_TILE + row * AS_STRIDE + col8);
            const unsigned gofs = (unsigned)((((m0 + row) << 10) + k0 + col8) * 2);
            async_b128(ldsb, gofs, Ab);
        }
        #pragma unroll
        for (int c = 0; c < 2; ++c) {
            const int chunk = tid * 2 + c;          // 0..511
            const int krow  = chunk >> 4;           // 0..31
            const int col8  = (chunk & 15) * 8;     // 0..120
            const unsigned gofs = (unsigned)(((((k0 + krow) << 10) + n0 + col8)) * 2);
            const unsigned lh = lds_addr32(BsH + buf * BS_TILE + krow * BS_STRIDE + col8);
            async_b128(lh, gofs, Bhb);
            const unsigned ll = lds_addr32(BsL + buf * BS_TILE + krow * BS_STRIDE + col8);
            async_b128(ll, gofs, Blb);
        }
    };

    stage(0, 0);

    for (int ks = 0; ks < ND / 32; ++ks) {
        const int cur   = ks & 1;
        const int knext = ((ks + 1) * 32) & (ND - 1);   // wraps -> harmless refill
        stage(knext, cur ^ 1);

        // async loads complete in order per wave: <=6 outstanding => prev slab done
        asm volatile("s_wait_asynccnt 0x6" ::: "memory");
        __syncthreads();

        const f16* Ac = As  + cur * AS_TILE;
        const f16* Bh = BsH + cur * BS_TILE;
        const f16* Bl = BsL + cur * BS_TILE;

        // A fragments (ISA 7.12.2, 16-bit A 16x32):
        //   lane%16 = M row; elems 0..7 <- K=8h+0..7, elems 8..15 <- K=16+8h+0..7
        const int h8 = (lane >> 4) * 8;
        v16h af[2];
        #pragma unroll
        for (int mi = 0; mi < 2; ++mi) {
            const int am = wm * 32 + mi * 16 + (lane & 15);
            af[mi] = cat8(*(const v8h*)(Ac + am * AS_STRIDE + h8),
                          *(const v8h*)(Ac + am * AS_STRIDE + 16 + h8));
        }

        // B fragment (32x16): lane%16 = N col; elems 0..7 <- K=8h+0..7, elems 8..15 <- K=16+8h+0..7
        #pragma unroll
        for (int f = 0; f < 4; ++f) {
            const int nb = wn * 64 + f * 16;
            const f16* bph = Bh + nb + (lane & 15);
            v16h bfh;
            #pragma unroll
            for (int s = 0; s < 16; ++s) bfh[s] = bph[(h8 + (s & 7) + (s >> 3) * 16) * BS_STRIDE];
            acc[0 * 4 + f] = wmma16(af[0], bfh, acc[0 * 4 + f]);
            acc[1 * 4 + f] = wmma16(af[1], bfh, acc[1 * 4 + f]);
            const f16* bpl = Bl + nb + (lane & 15);
            v16h bfl;
            #pragma unroll
            for (int s = 0; s < 16; ++s) bfl[s] = bpl[(h8 + (s & 7) + (s >> 3) * 16) * BS_STRIDE];
            acc[0 * 4 + f] = wmma16(af[0], bfl, acc[0 * 4 + f]);
            acc[1 * 4 + f] = wmma16(af[1], bfl, acc[1 * 4 + f]);
        }
        __syncthreads();   // all waves done reading `cur` before it is refilled
    }

    // epilogue: C/D layout -> VGPR i holds M = i + 8*(lane/16), N = lane%16
    #pragma unroll
    for (int mi = 0; mi < 2; ++mi) {
        const int row0 = m0 + wm * 32 + mi * 16 + 8 * (lane >> 4);
        #pragma unroll
        for (int f = 0; f < 4; ++f) {
            const int col = n0 + wn * 64 + f * 16 + (lane & 15);
            const float bv = bias[col];
            #pragma unroll
            for (int i = 0; i < 8; ++i) {
                out[(size_t)(row0 + i) * ND + col] = acc[mi * 4 + f][i] + bv;
            }
        }
    }
}

// ---------------------------------------------------------------------------
// launch
// ---------------------------------------------------------------------------
extern "C" void kernel_launch(void* const* d_in, const int* in_sizes, int n_in,
                              void* d_out, int out_size, void* d_ws, size_t ws_size,
                              hipStream_t stream) {
    (void)in_sizes; (void)n_in; (void)out_size; (void)ws_size;

    const int*   tokens = (const int*)d_in[0];
    const float* tv     = (const float*)d_in[1];
    const float* W      = (const float*)d_in[2];
    const float* bias   = (const float*)d_in[3];
    float*       out    = (float*)d_out;

    char* ws = (char*)d_ws;
    f16* bundleH = (f16*)ws;                                        // 16 MB
    f16* WThi    = (f16*)(ws + (size_t)NROW * ND * sizeof(f16));    //  2 MB
    f16* WTlo    = (f16*)(ws + (size_t)NROW * ND * sizeof(f16)
                             + (size_t)ND * ND * sizeof(f16));      //  2 MB

    hdc_bundle_kernel<<<NROW, 256, 0, stream>>>(tokens, tv, bundleH);
    hdc_wsplit_kernel<<<(ND * ND) / 256, 256, 0, stream>>>(W, WThi, WTlo);
    hdc_gemm_kernel<<<64 * 8, 256, 0, stream>>>(bundleH, WThi, WTlo, bias, out);
}
